// MSA_35356170780803
// MI455X (gfx1250) — matrix operand order repaired and checked by reading, not compiled, not run on hardware
//
#include <hip/hip_runtime.h>

typedef __attribute__((ext_vector_type(4)))  float    v4f;
typedef __attribute__((ext_vector_type(8)))  float    v8f;
typedef __attribute__((ext_vector_type(4)))  int      v4i;
typedef __attribute__((ext_vector_type(8)))  _Float16 v8h;
typedef __attribute__((ext_vector_type(16))) _Float16 v16h;

#define B_   8
#define C_   256
#define P_   8
#define CH_  32
#define N_   16384   // 128*128
#define EPSF 1e-6f

// phase-1 tiling
#define MT     128     // rows of W per block (half of 256)
#define NT     64      // n columns per block
#define APITCH 40      // f16, 80B rows -> 16B aligned fragment chunks
#define BPITCH 264     // f16, 528B rows -> 16B aligned fragment chunks

// phase-2
#define CHUNK_N 512
#define NCHUNK  32     // 16384/512

__device__ __forceinline__ float softplus_f(float x) {
  return fmaxf(x, 0.0f) + log1pf(__expf(-fabsf(x)));
}

__device__ __forceinline__ v16h cat8(v8h a, v8h b) {
  return __builtin_shufflevector(a, b, 0,1,2,3,4,5,6,7,8,9,10,11,12,13,14,15);
}

// CDNA5 async global->LDS copy (ASYNCcnt path), with safe fallback.
// Signature per hipcc diagnostic: arg0 = v4i in global AS; 4 args total.
__device__ __forceinline__ void async_copy_b128(const float* g, float* l) {
#if __has_builtin(__builtin_amdgcn_global_load_async_to_lds_b128)
  typedef __attribute__((address_space(1))) v4i* gas_t;
  typedef __attribute__((address_space(3))) v4i* las_t;
  __builtin_amdgcn_global_load_async_to_lds_b128((gas_t)g, (las_t)l, 0, 0);
#else
  *(v4f*)l = *(const v4f*)g;
#endif
}

__device__ __forceinline__ void wait_async0() {
#if __has_builtin(__builtin_amdgcn_global_load_async_to_lds_b128)
#if __has_builtin(__builtin_amdgcn_s_wait_asynccnt)
  __builtin_amdgcn_s_wait_asynccnt(0);
#else
  asm volatile("s_wait_asynccnt 0" ::: "memory");
#endif
#endif
}

// ---------------------------------------------------------------------------
// Phase 1: fused QKV projection.  out[m,n] = act(W[m,:] @ X[:,n] + b[m])
// X panel [256 k][64 n] cached once in LDS (f16), reused for Wq/Wk/Wv.
// W tiles double-buffered in LDS: one barrier per k-step, load/WMMA overlap.
// ---------------------------------------------------------------------------
__global__ __launch_bounds__(256)
void qkv_proj_kernel(const float* __restrict__ x,
                     const float* __restrict__ Wq, const float* __restrict__ bq,
                     const float* __restrict__ Wk, const float* __restrict__ bk,
                     const float* __restrict__ Wv, const float* __restrict__ bv,
                     _Float16* __restrict__ qo, _Float16* __restrict__ ko,
                     _Float16* __restrict__ vo)
{
  __shared__ _Float16 ldsB[NT * BPITCH];        // X panel, [n][k], k = 0..255
  __shared__ _Float16 ldsA[2][MT * APITCH];     // W tile,  [m][k], double buffer
  __shared__ float    ldsBias[MT];

  const int tid  = threadIdx.x;
  const int lane = tid & 31;
  const int wave = tid >> 5;
  const int wm   = wave >> 2;     // 0..1  -> msubs wm*4..wm*4+3
  const int wn   = wave & 3;      // 0..3  -> nsub  wn
  const int lr   = lane & 15;
  const int hi   = lane >> 4;

  const int b  = blockIdx.z;
  const int mh = blockIdx.y;            // which 128-row half of W
  const int n0 = blockIdx.x * NT;

  // per-thread (row, col-group) for W tile staging
  const int wrow0 = tid >> 3;           // 0..31 (iter adds 32)
  const int wc4   = tid & 7;            // group of 4 k

  // ---- load X panel: x[b, k, n0..n0+63] -> ldsB[n][k] (transpose, fp32->f16)
  const float* xb = x + (size_t)b * C_ * N_;
  #pragma unroll
  for (int it = 0; it < 16; ++it) {
    int lin = tid + it * 256;           // 0..4095
    int k   = lin >> 4;                 // 0..255
    int ng  = lin & 15;                 // group of 4 n
    v4f d = *(const v4f*)(xb + (size_t)k * N_ + n0 + ng * 4);
    #pragma unroll
    for (int j = 0; j < 4; ++j)
      ldsB[(ng * 4 + j) * BPITCH + k] = (_Float16)d[j];
  }

  const float* Ws[3]   = {Wq, Wk, Wv};
  const float* bs[3]   = {bq, bk, bv};
  _Float16*    outs[3] = {qo, ko, vo};

  #pragma unroll
  for (int widx = 0; widx < 3; ++widx) {
    const float* Wm = Ws[widx];
    v8f acc[4];
    #pragma unroll
    for (int i = 0; i < 4; ++i) acc[i] = (v8f)0.0f;

    __syncthreads();                       // prior epilogue / buffers done
    if (tid < MT) ldsBias[tid] = bs[widx][mh * MT + tid];

    // preload k-tile 0 into buffer 0
    #pragma unroll
    for (int it = 0; it < 4; ++it) {
      int r = wrow0 + it * 32;
      v4f d = *(const v4f*)(Wm + (size_t)(mh * MT + r) * C_ + wc4 * 4);
      _Float16* pp = &ldsA[0][r * APITCH + wc4 * 4];
      pp[0] = (_Float16)d[0]; pp[1] = (_Float16)d[1];
      pp[2] = (_Float16)d[2]; pp[3] = (_Float16)d[3];
    }
    __syncthreads();

    #pragma unroll
    for (int kt = 0; kt < 8; ++kt) {
      const int cur = kt & 1;

      // stage next W tile from global while WMMAs below execute
      v4f stage[4];
      if (kt < 7) {
        #pragma unroll
        for (int it = 0; it < 4; ++it) {
          int r = wrow0 + it * 32;
          stage[it] = *(const v4f*)(Wm + (size_t)(mh * MT + r) * C_ +
                                    (kt + 1) * 32 + wc4 * 4);
        }
      }

      // A fragments: lane<16 -> K {0..7,16..23}; lane>=16 -> K {8..15,24..31}
      v16h afr[4];
      #pragma unroll
      for (int mi = 0; mi < 4; ++mi) {
        const _Float16* ap = &ldsA[cur][((wm * 4 + mi) * 16 + lr) * APITCH + hi * 8];
        afr[mi] = cat8(*(const v8h*)ap, *(const v8h*)(ap + 16));
      }
      // B fragment: lane<16 -> col lr, K {0..7,16..23}; lane>=16 -> col lr, K {8..15,24..31}
      const _Float16* bp = &ldsB[(wn * 16 + lr) * BPITCH + kt * 32 + hi * 8];
      v16h bfr = cat8(*(const v8h*)bp, *(const v8h*)(bp + 16));

      #pragma unroll
      for (int mi = 0; mi < 4; ++mi)
        acc[mi] = __builtin_amdgcn_wmma_f32_16x16x32_f16(
            false, afr[mi], false, bfr, (short)0, acc[mi], false, false);

      // commit staged tile to the other buffer
      if (kt < 7) {
        #pragma unroll
        for (int it = 0; it < 4; ++it) {
          int r = wrow0 + it * 32;
          _Float16* pp = &ldsA[1 - cur][r * APITCH + wc4 * 4];
          pp[0] = (_Float16)stage[it][0]; pp[1] = (_Float16)stage[it][1];
          pp[2] = (_Float16)stage[it][2]; pp[3] = (_Float16)stage[it][3];
        }
      }
      __syncthreads();
    }

    // epilogue: bias + activation, store f16
    _Float16* op = outs[widx] + (size_t)b * C_ * N_;
    const int ncol = n0 + wn * 16 + lr;
    #pragma unroll
    for (int mi = 0; mi < 4; ++mi) {
      #pragma unroll
      for (int r = 0; r < 8; ++r) {
        int ml = (wm * 4 + mi) * 16 + r + 8 * hi;   // C/D layout: M=r(+8), N=lane%16
        float val = acc[mi][r] + ldsBias[ml];
        if (widx < 2) val = softplus_f(val);
        op[(size_t)(mh * MT + ml) * N_ + ncol] = (_Float16)val;
      }
    }
  }
}

// ---------------------------------------------------------------------------
// Phase 2: per (b,p,chunk): partial kv[32][32] = sum_n k[m,n]*v[c,n], and
// partial ksum[m] = sum_n k[m,n].  WMMA with fragments loaded from global.
// ---------------------------------------------------------------------------
__global__ __launch_bounds__(256)
void kv_partial_kernel(const _Float16* __restrict__ ko,
                       const _Float16* __restrict__ vo,
                       float* __restrict__ pkv, float* __restrict__ pks)
{
  __shared__ float red[8][4][8][32];     // [wave][tile][vgpr r][lane]
  __shared__ float kred[8][2][32];       // [wave][mt][lane]

  const int tid  = threadIdx.x;
  const int lane = tid & 31;
  const int wave = tid >> 5;
  const int lr   = lane & 15;
  const int hi   = lane >> 4;
  const int chunk = blockIdx.x, p = blockIdx.y, b = blockIdx.z;
  const int bp = b * P_ + p;

  const _Float16* kk = ko + ((size_t)b * C_ + p * CH_) * N_;
  const _Float16* vv = vo + ((size_t)b * C_ + p * CH_) * N_;

  v8f acc[2][2];
  #pragma unroll
  for (int i = 0; i < 2; ++i)
    #pragma unroll
    for (int j = 0; j < 2; ++j) acc[i][j] = (v8f)0.0f;
  float ks[2] = {0.0f, 0.0f};

  for (int s = wave; s < 16; s += 8) {          // 16 k-steps of 32 n
    const int nb = chunk * CHUNK_N + s * 32;

    if (s + 8 < 16) {                           // prefetch next slab
      __builtin_prefetch(kk + (size_t)lr * N_ + nb + 256, 0, 3);
      __builtin_prefetch(vv + (size_t)lr * N_ + nb + 256, 0, 3);
    }

    v16h afr[2], bfr[2];
    #pragma unroll
    for (int mt = 0; mt < 2; ++mt) {
      const _Float16* ap = kk + (size_t)(mt * 16 + lr) * N_ + nb + hi * 8;
      afr[mt] = cat8(*(const v8h*)ap, *(const v8h*)(ap + 16));
      #pragma unroll
      for (int e = 0; e < 16; ++e) ks[mt] += (float)afr[mt][e];  // ksum from A frag
    }
    #pragma unroll
    for (int ct = 0; ct < 2; ++ct) {
      const _Float16* bp2 = vv + (size_t)(ct * 16 + lr) * N_ + nb + hi * 8;
      bfr[ct] = cat8(*(const v8h*)bp2, *(const v8h*)(bp2 + 16));
    }
    #pragma unroll
    for (int mt = 0; mt < 2; ++mt)
      #pragma unroll
      for (int ct = 0; ct < 2; ++ct)
        acc[mt][ct] = __builtin_amdgcn_wmma_f32_16x16x32_f16(
            false, afr[mt], false, bfr[ct], (short)0, acc[mt][ct], false, false);
  }

  #pragma unroll
  for (int mt = 0; mt < 2; ++mt)
    #pragma unroll
    for (int ct = 0; ct < 2; ++ct)
      #pragma unroll
      for (int r = 0; r < 8; ++r)
        red[wave][mt * 2 + ct][r][lane] = acc[mt][ct][r];
  kred[wave][0][lane] = ks[0];
  kred[wave][1][lane] = ks[1];
  __syncthreads();

  float* outp = pkv + ((size_t)bp * NCHUNK + chunk) * (CH_ * CH_);
  #pragma unroll
  for (int i = 0; i < 4; ++i) {
    int pos  = tid + i * 256;                 // 0..1023
    int tile = pos >> 8, r = (pos >> 5) & 7, l = pos & 31;
    float s = 0.0f;
    #pragma unroll
    for (int w = 0; w < 8; ++w) s += red[w][tile][r][l];
    int mt = tile >> 1, ct = tile & 1;
    int m  = mt * 16 + r + 8 * (l >> 4);
    int c  = ct * 16 + (l & 15);
    outp[m * CH_ + c] = s;
  }
  if (tid < CH_) {
    int mt = tid >> 4, lrr = tid & 15;
    float s = 0.0f;
    #pragma unroll
    for (int w = 0; w < 8; ++w) s += kred[w][mt][lrr] + kred[w][mt][lrr + 16];
    pks[((size_t)bp * NCHUNK + chunk) * CH_ + tid] = s;
  }
}

// ---------------------------------------------------------------------------
// Phase 2b: fixed-order reduction of chunk partials (deterministic).
// ---------------------------------------------------------------------------
__global__ __launch_bounds__(256)
void kv_reduce_kernel(const float* __restrict__ pkv, const float* __restrict__ pks,
                      float* __restrict__ kvf, float* __restrict__ ksf)
{
  const int bp = blockIdx.x, tid = threadIdx.x;
  #pragma unroll
  for (int i = 0; i < 4; ++i) {
    int idx = tid + i * 256;
    float s = 0.0f;
    for (int c = 0; c < NCHUNK; ++c)
      s += pkv[((size_t)bp * NCHUNK + c) * (CH_ * CH_) + idx];
    kvf[(size_t)bp * (CH_ * CH_) + idx] = s;
  }
  if (tid < CH_) {
    float s = 0.0f;
    for (int c = 0; c < NCHUNK; ++c)
      s += pks[((size_t)bp * NCHUNK + c) * CH_ + tid];
    ksf[(size_t)bp * CH_ + tid] = s;
  }
}

// ---------------------------------------------------------------------------
// Phase 3: out[c,n] = (q[:,n] . kv[:,c]) / (q[:,n] . (ksum + eps))
// kv tile (4 KB) fetched with one async global->LDS B128 op per lane.
// ---------------------------------------------------------------------------
__global__ __launch_bounds__(256)
void out_kernel(const _Float16* __restrict__ qo, const float* __restrict__ kvf,
                const float* __restrict__ ksf, float* __restrict__ out)
{
  __shared__ float skv[CH_ * CH_];
  __shared__ float sks[CH_];
  const int tid = threadIdx.x;
  const int p = blockIdx.y, b = blockIdx.z;
  const int bp = b * P_ + p;

  // 256 lanes x 16B = entire 32x32 fp32 kv tile in one async op
  async_copy_b128(kvf + (size_t)bp * (CH_ * CH_) + tid * 4, &skv[tid * 4]);
  if (tid < CH_) sks[tid] = ksf[(size_t)bp * CH_ + tid] + EPSF;
  wait_async0();
  __syncthreads();

  const int n = blockIdx.x * 256 + tid;
  const _Float16* qp = qo + ((size_t)b * C_ + p * CH_) * N_ + n;
  float q[CH_];
  float denom = 0.0f;
  #pragma unroll
  for (int m = 0; m < CH_; ++m) {
    q[m] = (float)qp[(size_t)m * N_];
    denom += q[m] * sks[m];
  }
  const float norm = 1.0f / denom;
  float* op = out + ((size_t)b * C_ + p * CH_) * N_ + n;
  #pragma unroll
  for (int c = 0; c < CH_; ++c) {
    float s = 0.0f;
    #pragma unroll
    for (int m = 0; m < CH_; ++m) s += q[m] * skv[m * CH_ + c];
    op[(size_t)c * N_] = norm * s;
  }
}

// ---------------------------------------------------------------------------
extern "C" void kernel_launch(void* const* d_in, const int* in_sizes, int n_in,
                              void* d_out, int out_size, void* d_ws, size_t ws_size,
                              hipStream_t stream)
{
  (void)in_sizes; (void)n_in; (void)out_size; (void)ws_size;
  const float* x  = (const float*)d_in[0];
  const float* Wq = (const float*)d_in[1];
  const float* bq = (const float*)d_in[2];
  const float* Wk = (const float*)d_in[3];
  const float* bk = (const float*)d_in[4];
  const float* Wv = (const float*)d_in[5];
  const float* bv = (const float*)d_in[6];
  float* out = (float*)d_out;

  const size_t qkv_elems = (size_t)B_ * C_ * N_;     // 33.5M f16 each
  _Float16* qo = (_Float16*)d_ws;
  _Float16* ko = qo + qkv_elems;
  _Float16* vo = ko + qkv_elems;
  float* pkv = (float*)(vo + qkv_elems);                       // [64][32][1024]
  float* pks = pkv + (size_t)B_ * P_ * NCHUNK * CH_ * CH_;     // [64][32][32]
  float* kvf = pks + (size_t)B_ * P_ * NCHUNK * CH_;           // [64][1024]
  float* ksf = kvf + (size_t)B_ * P_ * CH_ * CH_;              // [64][32]

  qkv_proj_kernel<<<dim3(N_ / NT, 2, B_), 256, 0, stream>>>(
      x, Wq, bq, Wk, bk, Wv, bv, qo, ko, vo);
  kv_partial_kernel<<<dim3(NCHUNK, P_, B_), 256, 0, stream>>>(ko, vo, pkv, pks);
  kv_reduce_kernel<<<dim3(B_ * P_), 256, 0, stream>>>(pkv, pks, kvf, ksf);
  out_kernel<<<dim3(N_ / 256, P_, B_), 256, 0, stream>>>(qo, kvf, ksf, out);
}
